// TiledFlashAttention_8581344657498
// MI455X (gfx1250) — compile-verified, matrix operand order checked by reading, not run on hardware
//
#include <hip/hip_runtime.h>
#include <hip/hip_bf16.h>

#if defined(__has_builtin)
#  if __has_builtin(__builtin_amdgcn_global_load_async_to_lds_b128)
#    define USE_ASYNC_LDS 1
#  endif
#endif
#ifndef USE_ASYNC_LDS
#  define USE_ASYNC_LDS 0
#endif

typedef __attribute__((ext_vector_type(16))) __bf16 v16bf;
typedef __attribute__((ext_vector_type(8)))  __bf16 v8bf;
typedef __attribute__((ext_vector_type(8)))  float  v8f;
typedef __attribute__((ext_vector_type(4)))  int    v4i;

#define B_  2
#define T_  2048
#define D_  1024
#define H_  16
#define HD_ 64
#define BT_ (B_*T_)   // 4096

// ---- 16-byte global -> LDS copy: async DMA path (ASYNCcnt) or fallback ----
__device__ __forceinline__ void cp16(const __bf16* g, __bf16* l) {
#if USE_ASYNC_LDS
  __builtin_amdgcn_global_load_async_to_lds_b128(
      (__attribute__((address_space(1))) v4i*)g,
      (__attribute__((address_space(3))) v4i*)l, 0, 0);
#else
  *(v8bf*)l = *(const v8bf*)g;
#endif
}
__device__ __forceinline__ void cp_wait() {
#if USE_ASYNC_LDS
#  if __has_builtin(__builtin_amdgcn_s_wait_asynccnt)
  __builtin_amdgcn_s_wait_asynccnt(0);
#  else
  asm volatile("s_wait_asynccnt 0" ::: "memory");
#  endif
#endif
}

// WMMA A/B fragment load (16x32 bf16, ISA 7.12.2 layout). p is already offset
// by the lane's K-half (0 or 8); reads K {0..7} and {16..23} relative to p.
__device__ __forceinline__ v16bf load_frag(const __bf16* p) {
  v8bf lo = *(const v8bf*)(p);
  v8bf hi = *(const v8bf*)(p + 16);
  return __builtin_shufflevector(lo, hi, 0,1,2,3,4,5,6,7,8,9,10,11,12,13,14,15);
}

__device__ __forceinline__ v8f wmma_bf16(v16bf a, v16bf b, v8f c) {
  return __builtin_amdgcn_wmma_f32_16x16x32_bf16(false, a, false, b,
                                                 (short)0, c, false, false);
}

// Reductions across the 16 lanes of a wave half (masks 1,2,4,8 stay in-half).
__device__ __forceinline__ float hmax16(float v) {
  v = fmaxf(v, __shfl_xor(v, 1, 32));
  v = fmaxf(v, __shfl_xor(v, 2, 32));
  v = fmaxf(v, __shfl_xor(v, 4, 32));
  v = fmaxf(v, __shfl_xor(v, 8, 32));
  return v;
}
__device__ __forceinline__ float hsum16(float v) {
  v += __shfl_xor(v, 1, 32);
  v += __shfl_xor(v, 2, 32);
  v += __shfl_xor(v, 4, 32);
  v += __shfl_xor(v, 8, 32);
  return v;
}

// ---------------- conversion kernels ----------------
__global__ void k_cvt(const float* __restrict__ s, __bf16* __restrict__ d, int n) {
  int i = blockIdx.x * blockDim.x + threadIdx.x;
  if (i < n) d[i] = (__bf16)s[i];
}
// src [rows][cols] f32  ->  dst [cols][rows] bf16  (transpose)
__global__ void k_cvt_t(const float* __restrict__ s, __bf16* __restrict__ d,
                        int rows, int cols) {
  int j = blockIdx.x * blockDim.x + threadIdx.x;
  if (j >= rows * cols) return;
  int n = j / rows, k = j - n * rows;
  d[j] = (__bf16)s[k * cols + n];
}

// ---------------- shared GEMM core ----------------
// Block = 8 waves computes a 256x64 tile of C = A[.,1024] @ Bt[.,1024]^T.
// Each wave owns a 32x64 sub-tile (2 A-frags x 4 B-frags = 8 WMMAs/step).
// A/B K-step tiles are staged in double-buffered LDS via async copies so the
// DMA for step i+1 overlaps the WMMAs of step i.
#define KSTEPS (D_/32)

__device__ __forceinline__ void stage_ab(const __bf16* gA, const __bf16* gB,
                                         int k0, __bf16* as, __bf16* bs, int tid) {
  int r = tid >> 2, o = (tid & 3) << 3;          // 4 threads per 32-elem row
#pragma unroll
  for (int j = 0; j < 4; ++j)                    // A rows r, r+64, r+128, r+192
    cp16(gA + (size_t)(r + j * 64) * D_ + k0 + o, as + (tid + j * 256) * 8);
  cp16(gB + (size_t)r * D_ + k0 + o, bs + tid * 8);   // B rows 0..63
}

__device__ __forceinline__ void gemm_core(const __bf16* __restrict__ A,
                                          const __bf16* __restrict__ Bt,
                                          int rowBase, int colBase,
                                          __bf16* As, __bf16* Bs, v8f acc[2][4]) {
  const int tid  = threadIdx.x;
  const int wib  = tid >> 5;
  const int lane = tid & 31;
  const int row  = lane & 15;
  const int kh   = (lane >> 4) << 3;
  const __bf16* gA = A  + (size_t)rowBase * D_;
  const __bf16* gB = Bt + (size_t)colBase * D_;

  stage_ab(gA, gB, 0, As, Bs, tid);
  for (int i = 0; i < KSTEPS; ++i) {
    __bf16* as = As + (i & 1) * (256 * 32);
    __bf16* bs = Bs + (i & 1) * (64 * 32);
    cp_wait();
    __syncthreads();
    if (i + 1 < KSTEPS)
      stage_ab(gA, gB, (i + 1) * 32,
               As + ((i + 1) & 1) * (256 * 32),
               Bs + ((i + 1) & 1) * (64 * 32), tid);
    // hoist all fragment loads so ds_loads issue as one clause
    v16bf a0 = load_frag(as + (wib * 32 + row) * 32 + kh);
    v16bf a1 = load_frag(as + (wib * 32 + 16 + row) * 32 + kh);
    v16bf b0 = load_frag(bs + (0 * 16 + row) * 32 + kh);
    v16bf b1 = load_frag(bs + (1 * 16 + row) * 32 + kh);
    v16bf b2 = load_frag(bs + (2 * 16 + row) * 32 + kh);
    v16bf b3 = load_frag(bs + (3 * 16 + row) * 32 + kh);
    acc[0][0] = wmma_bf16(a0, b0, acc[0][0]);
    acc[0][1] = wmma_bf16(a0, b1, acc[0][1]);
    acc[0][2] = wmma_bf16(a0, b2, acc[0][2]);
    acc[0][3] = wmma_bf16(a0, b3, acc[0][3]);
    acc[1][0] = wmma_bf16(a1, b0, acc[1][0]);
    acc[1][1] = wmma_bf16(a1, b1, acc[1][1]);
    acc[1][2] = wmma_bf16(a1, b2, acc[1][2]);
    acc[1][3] = wmma_bf16(a1, b3, acc[1][3]);
  }
}

// ---------------- QKV projection GEMM ----------------
__global__ void __launch_bounds__(256)
k_qkv_gemm(const __bf16* __restrict__ X, const __bf16* __restrict__ WT,
           __bf16* __restrict__ Qb, __bf16* __restrict__ Kb,
           __bf16* __restrict__ VTb) {
  __shared__ __align__(16) __bf16 As[2 * 256 * 32];   // 32 KB
  __shared__ __align__(16) __bf16 Bs[2 * 64 * 32];    // 8 KB
  const int MT = BT_ / 256;             // 16 row-tiles
  const int tm = blockIdx.x % MT;
  const int tn = blockIdx.x / MT;       // 0..47
  v8f acc[2][4] = {};
  gemm_core(X, WT, tm * 256, tn * 64, As, Bs, acc);

  const int wib = threadIdx.x >> 5, lane = threadIdx.x & 31;
  const int col = lane & 15;
#pragma unroll
  for (int g = 0; g < 2; ++g) {
    const int mb = tm * 256 + wib * 32 + g * 16 + ((lane >> 4) << 3);
#pragma unroll
    for (int f = 0; f < 4; ++f)
#pragma unroll
      for (int r = 0; r < 8; ++r) {
        int gr = mb + r;
        int gc = tn * 64 + f * 16 + col;
        int which = gc >> 10, dc = gc & (D_ - 1);
        int hh = dc >> 6, hi = dc & (HD_ - 1);
        int bb = gr >> 11, tr = gr & (T_ - 1);
        __bf16 v = (__bf16)acc[g][f][r];
        if (which == 0)      Qb[((size_t)(bb * H_ + hh) * T_ + tr) * HD_ + hi] = v;
        else if (which == 1) Kb[((size_t)(bb * H_ + hh) * T_ + tr) * HD_ + hi] = v;
        else                 VTb[((size_t)(bb * H_ + hh) * HD_ + hi) * T_ + tr] = v;
      }
  }
}

// ---------------- causal flash attention ----------------
// Block = 8 waves = 8 consecutive 16-row q-tiles of one (b,h); K/V 32-key
// tiles are shared by all waves and staged via double-buffered async copies.
__global__ void __launch_bounds__(256)
k_attn(const __bf16* __restrict__ Qb, const __bf16* __restrict__ Kb,
       const __bf16* __restrict__ VTb, __bf16* __restrict__ Ob) {
  __shared__ __align__(16) __bf16 Ksm[2 * 32 * 64];   // [key][hd]
  __shared__ __align__(16) __bf16 Vsm[2 * 64 * 32];   // [hd][key]
  __shared__ __align__(16) __bf16 Psm[8 * 16 * 32];   // per-wave P tile
  const int tid  = threadIdx.x;
  const int wib  = tid >> 5;
  const int lane = tid & 31;
  const int qblk = blockIdx.x & 15;
  const int h    = (blockIdx.x >> 4) & 15;
  const int b    = blockIdx.x >> 8;
  const int qt   = qblk * 8 + wib;
  const int row  = lane & 15;
  const int kh   = (lane >> 4) << 3;
  const int col  = lane & 15;
  const float NEG = -3.0e38f;

  const __bf16* Q  = Qb  + (size_t)(b * H_ + h) * T_ * HD_;
  const __bf16* K  = Kb  + (size_t)(b * H_ + h) * T_ * HD_;
  const __bf16* Vt = VTb + (size_t)(b * H_ + h) * HD_ * T_;
  __bf16* pt = Psm + wib * (16 * 32);

  const __bf16* qrow = Q + (size_t)(qt * 16 + row) * HD_;
  v16bf aq0 = load_frag(qrow + kh);
  v16bf aq1 = load_frag(qrow + 32 + kh);

  float m_i[8], l_i[8];
#pragma unroll
  for (int r = 0; r < 8; ++r) { m_i[r] = NEG; l_i[r] = 0.f; }
  v8f o[4] = {};

  const int qmax = qt * 16 + 15;
  const int mb   = qt * 16 + ((lane >> 4) << 3);
  const int nsteps = (qblk + 1) * 4;          // keys 0 .. qblk*128+127

  auto stage_kv = [&](int kt, int buf) {
    __bf16* ks = Ksm + buf * (32 * 64);
    __bf16* vs = Vsm + buf * (64 * 32);
    int kr = tid >> 3, ko = (tid & 7) << 3;   // K tile: 32 rows x 64 elems
    cp16(K + (size_t)(kt + kr) * HD_ + ko, ks + tid * 8);
    int vr = tid >> 2, vo = (tid & 3) << 3;   // V tile: 64 rows x 32 elems
    cp16(Vt + (size_t)vr * T_ + kt + vo, vs + tid * 8);
  };

  stage_kv(0, 0);
  for (int i = 0; i < nsteps; ++i) {
    const int kt = i * 32;
    cp_wait();
    __syncthreads();
    if (i + 1 < nsteps) stage_kv(kt + 32, (i + 1) & 1);

    if (kt <= qmax) {               // wave-uniform: EXEC stays all-ones
      __bf16* ks = Ksm + (i & 1) * (32 * 64);
      __bf16* vs = Vsm + (i & 1) * (64 * 32);

      // hoist K fragments, then S WMMAs
      v16bf bk0 = load_frag(ks + row * 64 + kh);
      v16bf bk1 = load_frag(ks + row * 64 + 32 + kh);
      v16bf bk2 = load_frag(ks + (16 + row) * 64 + kh);
      v16bf bk3 = load_frag(ks + (16 + row) * 64 + 32 + kh);
      v8f s0 = {}, s1 = {};
      s0 = wmma_bf16(aq0, bk0, s0);
      s0 = wmma_bf16(aq1, bk1, s0);
      s1 = wmma_bf16(aq0, bk2, s1);
      s1 = wmma_bf16(aq1, bk3, s1);

      // V fragments issued now: their LDS latency hides under the softmax
      v16bf bv0 = load_frag(vs + (0 * 16 + row) * 32 + kh);
      v16bf bv1 = load_frag(vs + (1 * 16 + row) * 32 + kh);
      v16bf bv2 = load_frag(vs + (2 * 16 + row) * 32 + kh);
      v16bf bv3 = load_frag(vs + (3 * 16 + row) * 32 + kh);

#pragma unroll
      for (int r = 0; r < 8; ++r) {
        int qg = mb + r;
        float v0 = (kt + col      <= qg) ? s0[r] * 0.125f : NEG;
        float v1 = (kt + 16 + col <= qg) ? s1[r] * 0.125f : NEG;
        float mx   = hmax16(fmaxf(v0, v1));
        float mnew = fmaxf(m_i[r], mx);
        float corr = __expf(m_i[r] - mnew);
        float p0 = __expf(v0 - mnew);
        float p1 = __expf(v1 - mnew);
        float ps = hsum16(p0 + p1);
        m_i[r] = mnew;
        l_i[r] = l_i[r] * corr + ps;
#pragma unroll
        for (int f = 0; f < 4; ++f) o[f][r] = o[f][r] * corr;
        s0[r] = p0; s1[r] = p1;
      }

      // stage P (16x32) per-wave in LDS as bf16, reload as A-fragment
#pragma unroll
      for (int r = 0; r < 8; ++r) {
        int m = ((lane >> 4) << 3) + r;
        pt[m * 32 + col]      = (__bf16)s0[r];
        pt[m * 32 + 16 + col] = (__bf16)s1[r];
      }
      __builtin_amdgcn_wave_barrier();
      v16bf ap = load_frag(pt + row * 32 + kh);
      __builtin_amdgcn_wave_barrier();

      o[0] = wmma_bf16(ap, bv0, o[0]);
      o[1] = wmma_bf16(ap, bv1, o[1]);
      o[2] = wmma_bf16(ap, bv2, o[2]);
      o[3] = wmma_bf16(ap, bv3, o[3]);
    }
  }

  // normalize and store O as bf16 in [b, t, d] layout
#pragma unroll
  for (int f = 0; f < 4; ++f)
#pragma unroll
    for (int r = 0; r < 8; ++r) {
      int tq = mb + r;
      float val = o[f][r] / l_i[r];
      Ob[((size_t)(b * T_) + tq) * D_ + h * HD_ + f * 16 + col] = (__bf16)val;
    }
}

// ---------------- output projection GEMM ----------------
__global__ void __launch_bounds__(256)
k_out_gemm(const __bf16* __restrict__ O, const __bf16* __restrict__ WT,
           float* __restrict__ out) {
  __shared__ __align__(16) __bf16 As[2 * 256 * 32];
  __shared__ __align__(16) __bf16 Bs[2 * 64 * 32];
  const int MT = BT_ / 256;
  const int tm = blockIdx.x % MT;
  const int tn = blockIdx.x / MT;       // 0..15
  v8f acc[2][4] = {};
  gemm_core(O, WT, tm * 256, tn * 64, As, Bs, acc);

  const int wib = threadIdx.x >> 5, lane = threadIdx.x & 31;
  const int col = lane & 15;
#pragma unroll
  for (int g = 0; g < 2; ++g) {
    const int mb = tm * 256 + wib * 32 + g * 16 + ((lane >> 4) << 3);
#pragma unroll
    for (int f = 0; f < 4; ++f)
#pragma unroll
      for (int r = 0; r < 8; ++r)
        out[(size_t)(mb + r) * D_ + tn * 64 + f * 16 + col] = acc[g][f][r];
  }
}

extern "C" void kernel_launch(void* const* d_in, const int* in_sizes, int n_in,
                              void* d_out, int out_size, void* d_ws, size_t ws_size,
                              hipStream_t stream) {
  const float* x    = (const float*)d_in[0];
  const float* wqkv = (const float*)d_in[1];
  const float* wout = (const float*)d_in[2];
  float* out = (float*)d_out;

  char* ws = (char*)d_ws;
  size_t off = 0;
  auto alloc = [&](size_t bytes) -> __bf16* {
    __bf16* p = (__bf16*)(ws + off);
    off += (bytes + 255) & ~(size_t)255;
    return p;
  };
  __bf16* xb  = alloc((size_t)BT_ * D_ * 2);       // x in bf16
  __bf16* wqT = alloc((size_t)3 * D_ * D_ * 2);    // w_qkv^T bf16 [3072][1024]
  __bf16* woT = alloc((size_t)D_ * D_ * 2);        // w_out^T bf16 [1024][1024]
  __bf16* qb  = alloc((size_t)BT_ * D_ * 2);       // Q  [b,h,t,hd]
  __bf16* kb  = alloc((size_t)BT_ * D_ * 2);       // K  [b,h,t,hd]
  __bf16* vtb = alloc((size_t)BT_ * D_ * 2);       // V^T[b,h,hd,t]
  __bf16* ob  = alloc((size_t)BT_ * D_ * 2);       // O  [b,t,d]
  (void)ws_size; (void)in_sizes; (void)n_in; (void)out_size;

  int n1 = BT_ * D_;
  k_cvt<<<(n1 + 255) / 256, 256, 0, stream>>>(x, xb, n1);
  int n2 = D_ * 3 * D_;
  k_cvt_t<<<(n2 + 255) / 256, 256, 0, stream>>>(wqkv, wqT, D_, 3 * D_);
  int n3 = D_ * D_;
  k_cvt_t<<<(n3 + 255) / 256, 256, 0, stream>>>(wout, woT, D_, D_);

  k_qkv_gemm<<<(BT_ / 256) * (3 * D_ / 64), 256, 0, stream>>>(xb, wqT, qb, kb, vtb);
  k_attn<<<B_ * H_ * (T_ / 128), 256, 0, stream>>>(qb, kb, vtb, ob);
  k_out_gemm<<<(BT_ / 256) * (D_ / 64), 256, 0, stream>>>(ob, woT, out);
}
